// PCN_25340307046598
// MI455X (gfx1250) — matrix operand order repaired and checked by reading, not compiled, not run on hardware
//
#include <hip/hip_runtime.h>

// MI455X (gfx1250) predictive-coding network relaxation.
// Reference output == s3 (c3 is never updated) -> d_out is a bandwidth copy;
// the WMMA pipeline implements the real PCN iteration:
//   5 GEMMs/cycle, f16 operands (pre-materialized, relu fused), f32 accum,
//   B panels staged through LDS via global_load_async_to_lds_b128 (ASYNCcnt),
//   double buffered; A fragments software-pipelined in registers.

typedef __attribute__((ext_vector_type(16))) _Float16     v16h;
typedef __attribute__((ext_vector_type(8)))  _Float16     v8h;
typedef __attribute__((ext_vector_type(8)))  float        v8f;
typedef __attribute__((ext_vector_type(4)))  unsigned int v4u;
typedef __attribute__((ext_vector_type(4)))  int          v4i;

#define HID0 1024
#define HID1 2048
#define HID2 2048
#define HID3 1024
#define BATCH 4096
#define NCYCLES 10
#define LRATE 0.1f

// LDS B panel: 64 rows (n) x 32 cols (k) of f16, rows padded 32->40 elements
#define SB_STRIDE 40
#define SB_ELEMS  (64 * SB_STRIDE)

// ---- async global->LDS staging (gfx1250 ASYNCcnt path) with sync fallback ----
#if __has_builtin(__builtin_amdgcn_global_load_async_to_lds_b128)
#define PCN_ASYNC 1
#else
#define PCN_ASYNC 0
#endif

__device__ __forceinline__ void stage16(const _Float16* g, _Float16* l) {
#if PCN_ASYNC
    __builtin_amdgcn_global_load_async_to_lds_b128(
        (__attribute__((address_space(1))) v4i*)g,
        (__attribute__((address_space(3))) v4i*)l, 0, 0);
#else
    *(v4u*)l = *(const v4u*)g;
#endif
}

__device__ __forceinline__ void stage_wait() {
#if PCN_ASYNC
#if __has_builtin(__builtin_amdgcn_s_wait_asynccnt)
    __builtin_amdgcn_s_wait_asynccnt(0);
#else
    asm volatile("s_wait_asynccnt 0x0" ::: "memory");
#endif
#endif
}

// ---------------- elementwise helpers ----------------
__global__ void k_copy_f32(const float* __restrict__ s, float* __restrict__ d, int n) {
    int i = blockIdx.x * blockDim.x + threadIdx.x;
    if (i < n) d[i] = s[i];
}
__global__ void k_f32_to_f16(const float* __restrict__ s, _Float16* __restrict__ d, int n) {
    int i = blockIdx.x * blockDim.x + threadIdx.x;
    if (i < n) d[i] = (_Float16)s[i];
}
__global__ void k_f32_to_f16_relu(const float* __restrict__ s, _Float16* __restrict__ d, int n) {
    int i = blockIdx.x * blockDim.x + threadIdx.x;
    if (i < n) d[i] = (_Float16)fmaxf(s[i], 0.0f);
}
// d[c*rows + r] = s[r*cols + c]  (W^T in f16 so e@W B-fragments are contiguous)
__global__ void k_f32_to_f16_t(const float* __restrict__ s, _Float16* __restrict__ d,
                               int rows, int cols) {
    int i = blockIdx.x * blockDim.x + threadIdx.x;
    if (i < rows * cols) {
        int r = i / cols;
        int c = i - r * cols;
        d[(size_t)c * rows + r] = (_Float16)s[i];
    }
}

// ---------------- WMMA mainloop ----------------
// Block = 256 threads = 8 waves -> 128x64 output tile; each wave owns 16x64.
// A fragment (16-bit A 16x32, ISA 7.12.2): lane m(+16): row m,
//   halves 0..7 = K kb+hi*8.., halves 8..15 = K kb+16+hi*8..
// B fragment: lane l holds column (l&15), same K order as A (kb+hi*8.., kb+16+hi*8..).
__device__ __forceinline__ v16h load_a(const _Float16* arow, int i) {
    v8h a0 = *(const v8h*)(arow + i * 32);
    v8h a1 = *(const v8h*)(arow + i * 32 + 16);
    return __builtin_shufflevector(a0, a1, 0, 1, 2, 3, 4, 5, 6, 7,
                                           8, 9, 10, 11, 12, 13, 14, 15);
}

// Batch all 8 ds_loads (one clause, one dscnt wait), then 4 back-to-back WMMAs.
__device__ __forceinline__ void wmma4(const _Float16* bp, v16h a, v8f acc[4]) {
    v16h b[4];
#pragma unroll
    for (int j = 0; j < 4; ++j) {
        v8h b0 = *(const v8h*)(bp + j * 16 * SB_STRIDE);
        v8h b1 = *(const v8h*)(bp + j * 16 * SB_STRIDE + 16);
        b[j] = __builtin_shufflevector(b0, b1, 0, 1, 2, 3, 4, 5, 6, 7,
                                               8, 9, 10, 11, 12, 13, 14, 15);
    }
#pragma unroll
    for (int j = 0; j < 4; ++j)
        acc[j] = __builtin_amdgcn_wmma_f32_16x16x32_f16(
            false, a, false, b[j], (short)0, acc[j], false, false);
}

__device__ __forceinline__ void gemm_mainloop(const _Float16* __restrict__ A,   // (M,K) f16
                                              const _Float16* __restrict__ Bw,  // (N,K) f16
                                              _Float16* sB, int mrow, int nbase,
                                              int hi, int lr, int K, v8f acc[4]) {
    const int t  = threadIdx.x;
    const int sr = t >> 2;             // staging row 0..63
    const int sc = (t & 3) * 8;        // staging chunk (8 f16 = 16 B)
    const _Float16* gsrc = Bw + (size_t)(nbase + sr) * K + sc;
    _Float16*       lrow = sB + sr * SB_STRIDE + sc;
    const _Float16* arow = A + (size_t)mrow * K + hi * 8;
    const _Float16* bcol = sB + lr * SB_STRIDE + hi * 8;
    const int nk = K >> 5;             // >= 32 for all our shapes

    stage16(gsrc, lrow);               // panel 0 -> buffer 0
    v16h a = load_a(arow, 0);
    stage_wait();
    __syncthreads();

    int i = 0;
    for (; i < nk - 1; ++i) {          // steady state: everything unconditional
        const int cur = i & 1;
        stage16(gsrc + (i + 1) * 32, lrow + (cur ^ 1) * SB_ELEMS);
        __builtin_prefetch(gsrc + (i + 2) * 32, 0, 1);   // speculative, in-ws
        v16h an = load_a(arow, i + 1);                    // pipeline next A
        wmma4(bcol + cur * SB_ELEMS, a, acc);
        a = an;
        stage_wait();
        __syncthreads();
    }
    wmma4(bcol + (i & 1) * SB_ELEMS, a, acc);            // peeled last K-step
}

// E = X - Ar @ W^T  with Ar = relu(state) already in f16.
// Writes fp32 E and/or f16 Eh (either may be null; uniform branches).
__global__ void __launch_bounds__(256) k_pred_err(
    const _Float16* __restrict__ Ar, const _Float16* __restrict__ W,
    const float* __restrict__ X, float* __restrict__ E, _Float16* __restrict__ Eh,
    int N, int K) {
    __shared__ _Float16 sB[2 * SB_ELEMS];
    const int ntg   = N >> 6;
    const int bm    = blockIdx.x / ntg;
    const int bn    = blockIdx.x - bm * ntg;
    const int wave  = threadIdx.x >> 5;
    const int lane  = threadIdx.x & 31;
    const int hi    = lane >> 4;
    const int lr    = lane & 15;
    const int mbase = bm * 128 + wave * 16;
    const int nbase = bn * 64;

    v8f z = {};
    v8f acc[4] = {z, z, z, z};
    gemm_mainloop(Ar, W, sB, mbase + lr, nbase, hi, lr, K, acc);

#pragma unroll
    for (int j = 0; j < 4; ++j) {
        int col = nbase + j * 16 + lr;
#pragma unroll
        for (int i = 0; i < 8; ++i) {
            size_t idx = (size_t)(mbase + hi * 8 + i) * N + col;
            float v = X[idx] - acc[j][i];
            if (E)  E[idx]  = v;
            if (Eh) Eh[idx] = (_Float16)v;
        }
    }
}

// G = Eh_prev @ W (Wt = W^T stored (N,K) f16);
// C += LR*(-Ecur + (C>0)*G);  Cr = relu(C_new) in f16 for next cycle's A.
__global__ void __launch_bounds__(256) k_grad_upd(
    const _Float16* __restrict__ Epre, const _Float16* __restrict__ Wt,
    const float* __restrict__ Ecur, float* __restrict__ C, _Float16* __restrict__ Cr,
    int N, int K) {
    __shared__ _Float16 sB[2 * SB_ELEMS];
    const int ntg   = N >> 6;
    const int bm    = blockIdx.x / ntg;
    const int bn    = blockIdx.x - bm * ntg;
    const int wave  = threadIdx.x >> 5;
    const int lane  = threadIdx.x & 31;
    const int hi    = lane >> 4;
    const int lr    = lane & 15;
    const int mbase = bm * 128 + wave * 16;
    const int nbase = bn * 64;

    v8f z = {};
    v8f acc[4] = {z, z, z, z};
    gemm_mainloop(Epre, Wt, sB, mbase + lr, nbase, hi, lr, K, acc);

#pragma unroll
    for (int j = 0; j < 4; ++j) {
        int col = nbase + j * 16 + lr;
#pragma unroll
        for (int i = 0; i < 8; ++i) {
            size_t idx = (size_t)(mbase + hi * 8 + i) * N + col;
            float c = C[idx];
            float e = Ecur[idx];
            float g = (c > 0.0f) ? acc[j][i] : 0.0f;
            float cn = c + LRATE * (g - e);
            C[idx]  = cn;
            Cr[idx] = (_Float16)fmaxf(cn, 0.0f);
        }
    }
}

// ---------------- host ----------------
extern "C" void kernel_launch(void* const* d_in, const int* in_sizes, int n_in,
                              void* d_out, int out_size, void* d_ws, size_t ws_size,
                              hipStream_t stream) {
    (void)in_sizes; (void)n_in; (void)out_size; (void)ws_size;

    const float* x0 = (const float*)d_in[0];   // (B, HID0) clamped layer 0
    const float* s1 = (const float*)d_in[1];   // (B, HID1)
    const float* s2 = (const float*)d_in[2];   // (B, HID2)
    const float* s3 = (const float*)d_in[3];   // (B, HID3) never updated -> output
    const float* W1 = (const float*)d_in[4];   // (HID0, HID1)
    const float* W2 = (const float*)d_in[5];   // (HID1, HID2)
    const float* W3 = (const float*)d_in[6];   // (HID2, HID3)

    char* p = (char*)d_ws;
    auto carve = [&](size_t bytes) -> void* {
        void* r = (void*)p;
        p += (bytes + 255) & ~(size_t)255;
        return r;
    };
    _Float16* W1h  = (_Float16*)carve((size_t)HID0 * HID1 * 2);  // (HID0,HID1)
    _Float16* W2h  = (_Float16*)carve((size_t)HID1 * HID2 * 2);
    _Float16* W3h  = (_Float16*)carve((size_t)HID2 * HID3 * 2);
    _Float16* W1ht = (_Float16*)carve((size_t)HID0 * HID1 * 2);  // (HID1,HID0)
    _Float16* W2ht = (_Float16*)carve((size_t)HID1 * HID2 * 2);  // (HID2,HID1)
    float*    c1   = (float*)carve((size_t)BATCH * HID1 * 4);
    float*    c2   = (float*)carve((size_t)BATCH * HID2 * 4);
    _Float16* c1r  = (_Float16*)carve((size_t)BATCH * HID1 * 2); // relu(c1) f16
    _Float16* c2r  = (_Float16*)carve((size_t)BATCH * HID2 * 2);
    _Float16* s3r  = (_Float16*)carve((size_t)BATCH * HID3 * 2); // relu(s3) f16 (const)
    _Float16* e0h  = (_Float16*)carve((size_t)BATCH * HID0 * 2);
    float*    e1   = (float*)carve((size_t)BATCH * HID1 * 4);
    _Float16* e1h  = (_Float16*)carve((size_t)BATCH * HID1 * 2);
    float*    e2   = (float*)carve((size_t)BATCH * HID2 * 4);

    const int TPB = 256;
    auto b1d = [](size_t n) { return (unsigned)((n + 255) / 256); };

    // one-time weight conversion / transposition
    k_f32_to_f16<<<b1d((size_t)HID0 * HID1), TPB, 0, stream>>>(W1, W1h, HID0 * HID1);
    k_f32_to_f16<<<b1d((size_t)HID1 * HID2), TPB, 0, stream>>>(W2, W2h, HID1 * HID2);
    k_f32_to_f16<<<b1d((size_t)HID2 * HID3), TPB, 0, stream>>>(W3, W3h, HID2 * HID3);
    k_f32_to_f16_t<<<b1d((size_t)HID0 * HID1), TPB, 0, stream>>>(W1, W1ht, HID0, HID1);
    k_f32_to_f16_t<<<b1d((size_t)HID1 * HID2), TPB, 0, stream>>>(W2, W2ht, HID1, HID2);

    // state init: fp32 copies + f16 relu views (kept fresh by fused epilogues)
    k_copy_f32<<<b1d((size_t)BATCH * HID1), TPB, 0, stream>>>(s1, c1, BATCH * HID1);
    k_copy_f32<<<b1d((size_t)BATCH * HID2), TPB, 0, stream>>>(s2, c2, BATCH * HID2);
    k_f32_to_f16_relu<<<b1d((size_t)BATCH * HID1), TPB, 0, stream>>>(s1, c1r, BATCH * HID1);
    k_f32_to_f16_relu<<<b1d((size_t)BATCH * HID2), TPB, 0, stream>>>(s2, c2r, BATCH * HID2);
    k_f32_to_f16_relu<<<b1d((size_t)BATCH * HID3), TPB, 0, stream>>>(s3, s3r, BATCH * HID3);

    // grid: (BATCH/128) * (N/64) blocks of 256 threads (exact tiling)
    const unsigned g_n1024 = (BATCH / 128) * (HID0 / 64);  // 512
    const unsigned g_n2048 = (BATCH / 128) * (HID1 / 64);  // 1024

    for (int cyc = 0; cyc < NCYCLES; ++cyc) {
        // e0 = x0 - relu(c1)@W1^T   (N=HID0, K=HID1) -> f16 only (used as A of upd1)
        k_pred_err<<<g_n1024, TPB, 0, stream>>>(c1r, W1h, x0, (float*)nullptr, e0h, HID0, HID1);
        // e1 = c1 - relu(c2)@W2^T   (N=HID1, K=HID2) -> fp32 (Ecur of upd1) + f16 (A of upd2)
        k_pred_err<<<g_n2048, TPB, 0, stream>>>(c2r, W2h, c1, e1, e1h, HID1, HID2);
        // e2 = c2 - relu(s3)@W3^T   (N=HID2, K=HID3) -> fp32 only (Ecur of upd2)
        k_pred_err<<<g_n2048, TPB, 0, stream>>>(s3r, W3h, c2, e2, (_Float16*)nullptr, HID2, HID3);
        // c1 += LR*(-e1 + (c1>0)*(e0@W1));  c1r = relu(c1) f16   (N=HID1, K=HID0)
        k_grad_upd<<<g_n2048, TPB, 0, stream>>>(e0h, W1ht, e1, c1, c1r, HID1, HID0);
        // c2 += LR*(-e2 + (c2>0)*(e1@W2));  c2r = relu(c2) f16   (N=HID2, K=HID1)
        k_grad_upd<<<g_n2048, TPB, 0, stream>>>(e1h, W2ht, e2, c2, c2r, HID2, HID1);
    }

    // c3 is never updated by the reference: output = s3 (bandwidth-bound copy)
    k_copy_f32<<<b1d((size_t)BATCH * HID3), TPB, 0, stream>>>(s3, (float*)d_out, BATCH * HID3);
}
